// ExoplanetGNN_50508815401658
// MI455X (gfx1250) — matrix operand order repaired and checked by reading, not compiled, not run on hardware
//
#include <hip/hip_runtime.h>
#include <hip/hip_bf16.h>

typedef __attribute__((ext_vector_type(16))) _Float16 v16h;
typedef __attribute__((ext_vector_type(8)))  float    v8f;

// ---------------------------------------------------------------------------
// Pack a [Ksrc x 64] f32 weight (optionally summed with a second one) into the
// WMMA B-fragment layout for v_wmma_f32_16x16x32_f16, padded to kchunks*32 in K.
// Layout: out[((kc*4 + nt)*32 + lane)*16 + elem]
//   column n = nt*16 + (lane&15);  k = kc*32 + (elem>>3)*16 + (lane>>4)*8 + (elem&7)
// ---------------------------------------------------------------------------
__global__ void pack_weights_kernel(const float* __restrict__ W0,
                                    const float* __restrict__ Wadd,
                                    int Ksrc, int kchunks,
                                    _Float16* __restrict__ out) {
    int idx = blockIdx.x * blockDim.x + threadIdx.x;
    int total = kchunks * 4 * 32 * 16;
    if (idx >= total) return;
    int elem = idx & 15;
    int lane = (idx >> 4) & 31;
    int nt   = (idx >> 9) & 3;
    int kc   = idx >> 11;
    int n = nt * 16 + (lane & 15);
    int k = kc * 32 + (elem >> 3) * 16 + (lane >> 4) * 8 + (elem & 7);
    float v = 0.0f;
    if (k < Ksrc) {
        v = W0[k * 64 + n];
        if (Wadd) v += Wadd[k * 64 + n];
    }
    out[idx] = (_Float16)v;
}

__global__ void zero_kernel(float* __restrict__ p, int n) {
    int i = blockIdx.x * blockDim.x + threadIdx.x;
    if (i < n) p[i] = 0.0f;
}

__global__ void count_kernel(const int* __restrict__ dst, int E, float* __restrict__ cnt) {
    int e = blockIdx.x * blockDim.x + threadIdx.x;
    if (e < E) atomicAdd(cnt + dst[e], 1.0f);
}

__global__ void invert_kernel(float* __restrict__ c, int n) {
    int i = blockIdx.x * blockDim.x + threadIdx.x;
    if (i < n) c[i] = 1.0f / fmaxf(c[i], 1.0f);
}

// Edge-parallel gather + scatter-add: thread = (edge, 16B chunk of the 64-float row)
__global__ void scatter_add_kernel(const float* __restrict__ X,
                                   const int* __restrict__ src,
                                   const int* __restrict__ dst,
                                   int E, float* __restrict__ agg) {
    long idx = (long)blockIdx.x * blockDim.x + threadIdx.x;
    long total = (long)E * 16;
    if (idx >= total) return;
    int e = (int)(idx >> 4);
    int q = (int)(idx & 15);
    int s = src[e];
    int d = dst[e];
    const float4 v = *(const float4*)(X + (size_t)s * 64 + q * 4);
    float* a = agg + (size_t)d * 64 + q * 4;
    atomicAdd(a + 0, v.x);
    atomicAdd(a + 1, v.y);
    atomicAdd(a + 2, v.z);
    atomicAdd(a + 3, v.w);
}

// ---------------------------------------------------------------------------
// One operand pair's contribution to a 16x64 C tile.
// K is compile-time (16/32/64) -> all bounds fold, A row feed is aligned
// float4 (b128) loads; K=16 upper halves become literal zeros.
// A-fragment per lane (ISA 16-bit A 16x32 layout): row = lane&15, hi = lane>>4,
//   halves 0..7  -> K = kc*32 + hi*8 + j
//   halves 8..15 -> K = kc*32 + 16 + hi*8 + j
// ---------------------------------------------------------------------------
template<int K>
__device__ __forceinline__ void gemm_pair_acc(const float* __restrict__ A,
                                              const float* __restrict__ rs,
                                              const _Float16* __restrict__ P,
                                              int rowbase, int lane, v8f acc[4]) {
    const int r  = lane & 15;
    const int hi = lane >> 4;
    const float* rowp = A + (size_t)(rowbase + r) * K;
    const float scale = rs ? rs[rowbase + r] : 1.0f;
    const v16h* PB = (const v16h*)P;
    constexpr int KCH = (K + 31) / 32;
    #pragma unroll
    for (int kc = 0; kc < KCH; ++kc) {
        const int kb0 = kc * 32 + hi * 8;        // always < K for K in {16,32,64}
        const float4 lo0 = *(const float4*)(rowp + kb0);
        const float4 lo1 = *(const float4*)(rowp + kb0 + 4);
        v16h a;
        a[0] = (_Float16)(lo0.x * scale);
        a[1] = (_Float16)(lo0.y * scale);
        a[2] = (_Float16)(lo0.z * scale);
        a[3] = (_Float16)(lo0.w * scale);
        a[4] = (_Float16)(lo1.x * scale);
        a[5] = (_Float16)(lo1.y * scale);
        a[6] = (_Float16)(lo1.z * scale);
        a[7] = (_Float16)(lo1.w * scale);
        if (K >= (kc + 1) * 32) {                // compile-time per chunk
            const float4 hi0 = *(const float4*)(rowp + kb0 + 16);
            const float4 hi1 = *(const float4*)(rowp + kb0 + 20);
            a[8]  = (_Float16)(hi0.x * scale);
            a[9]  = (_Float16)(hi0.y * scale);
            a[10] = (_Float16)(hi0.z * scale);
            a[11] = (_Float16)(hi0.w * scale);
            a[12] = (_Float16)(hi1.x * scale);
            a[13] = (_Float16)(hi1.y * scale);
            a[14] = (_Float16)(hi1.z * scale);
            a[15] = (_Float16)(hi1.w * scale);
        } else {
            #pragma unroll
            for (int j = 8; j < 16; ++j) a[j] = (_Float16)0.0f;
        }
        #pragma unroll
        for (int nt = 0; nt < 4; ++nt) {
            v16h b = PB[(size_t)(kc * 4 + nt) * 32 + lane];
            acc[nt] = __builtin_amdgcn_wmma_f32_16x16x32_f16(
                /*neg_a=*/false, a, /*neg_b=*/false, b,
                /*c_mod=*/(short)0, acc[nt],
                /*reuse_a=*/false, /*reuse_b=*/false);
        }
    }
}

// ---------------------------------------------------------------------------
// Fused WMMA GEMM: C[M x 64] = act( outscale * ( sum_p (rs_p ⊙ A_p) @ W_p + b0 + b1 ) )
// Up to 3 operand pairs, selected at compile time (K_i == 0 -> unused).
// One wave per 16x64 C tile. M must be a multiple of 16 (holds: 500000, 200000),
// so waves exit uniformly and EXEC is all-ones at every WMMA.
// C/D f32 layout: elem i -> row = base + hi*8 + i, col = nt*16 + (lane&15)
// ---------------------------------------------------------------------------
template<int K0, int K1, int K2>
__global__ __launch_bounds__(128) void wmma_gemm_kernel(
    const float* __restrict__ A0, const float* __restrict__ rs0, const _Float16* __restrict__ P0,
    const float* __restrict__ A1, const float* __restrict__ rs1, const _Float16* __restrict__ P1,
    const float* __restrict__ A2, const float* __restrict__ rs2, const _Float16* __restrict__ P2,
    const float* __restrict__ bias0, const float* __restrict__ bias1,
    float outscale, int do_relu,
    float* __restrict__ C, int M) {

    int wave = (blockIdx.x * blockDim.x + threadIdx.x) >> 5;
    int rowbase = wave * 16;
    if (rowbase >= M) return;          // whole-wave uniform exit
    int lane = threadIdx.x & 31;

    v8f acc[4];
    #pragma unroll
    for (int nt = 0; nt < 4; ++nt) acc[nt] = (v8f){0.f,0.f,0.f,0.f,0.f,0.f,0.f,0.f};

    gemm_pair_acc<K0>(A0, rs0, P0, rowbase, lane, acc);
    if (K1 > 0) gemm_pair_acc<K1>(A1, rs1, P1, rowbase, lane, acc);
    if (K2 > 0) gemm_pair_acc<K2>(A2, rs2, P2, rowbase, lane, acc);

    const int coln = lane & 15;
    const int hi   = lane >> 4;
    #pragma unroll
    for (int nt = 0; nt < 4; ++nt) {
        const int col = nt * 16 + coln;
        float bb = 0.0f;
        if (bias0) bb += bias0[col];
        if (bias1) bb += bias1[col];
        #pragma unroll
        for (int i = 0; i < 8; ++i) {
            float v = (acc[nt][i] + bb) * outscale;
            if (do_relu) v = fmaxf(v, 0.0f);
            C[(size_t)(rowbase + hi * 8 + i) * 64 + col] = v;
        }
    }
}

// Readout head: out[i] = relu(hp[i]@W1 + b1) @ W2 + b2    (64 -> 32 -> 1)
__global__ void readout_kernel(const float* __restrict__ hp,
                               const float* __restrict__ W1, const float* __restrict__ b1,
                               const float* __restrict__ W2, const float* __restrict__ b2,
                               float* __restrict__ out, int n) {
    int i = blockIdx.x * blockDim.x + threadIdx.x;
    if (i >= n) return;
    const float* h = hp + (size_t)i * 64;
    float t[32];
    #pragma unroll
    for (int j = 0; j < 32; ++j) t[j] = b1[j];
    for (int k = 0; k < 64; ++k) {
        float hv = h[k];
        #pragma unroll
        for (int j = 0; j < 32; ++j) t[j] += hv * W1[k * 32 + j];
    }
    float acc = b2[0];
    #pragma unroll
    for (int j = 0; j < 32; ++j) acc += fmaxf(t[j], 0.0f) * W2[j];
    out[i] = acc;
}

// ---------------------------------------------------------------------------
// Host orchestration
// ---------------------------------------------------------------------------
static inline int cdiv(long a, long b) { return (int)((a + b - 1) / b); }

extern "C" void kernel_launch(void* const* d_in, const int* in_sizes, int n_in,
                              void* d_out, int out_size, void* d_ws, size_t ws_size,
                              hipStream_t stream) {
    const float* x_planet  = (const float*)d_in[0];
    const float* x_star    = (const float*)d_in[1];
    const int*   orb_src   = (const int*)d_in[2];
    const int*   orb_dst   = (const int*)d_in[3];
    const int*   host_src  = (const int*)d_in[4];
    const int*   host_dst  = (const int*)d_in[5];
    const int*   sib_src   = (const int*)d_in[6];
    const int*   sib_dst   = (const int*)d_in[7];
    const float* Wp = (const float*)d_in[8];
    const float* bp = (const float*)d_in[9];
    const float* Ws = (const float*)d_in[10];
    const float* bs = (const float*)d_in[11];
    const float* Wl = (const float*)d_in[12];   // [3,3,64,64]
    const float* bl = (const float*)d_in[13];   // [3,3,64]
    const float* Wr = (const float*)d_in[14];   // [3,3,64,64]
    const float* W1 = (const float*)d_in[15];   // [64,32]
    const float* b1 = (const float*)d_in[16];
    const float* W2 = (const float*)d_in[17];   // [32,1]
    const float* b2 = (const float*)d_in[18];

    const int NP = in_sizes[0] / 32;
    const int NS = in_sizes[1] / 16;
    const int EO = in_sizes[2];
    const int EH = in_sizes[4];
    const int ES = in_sizes[6];
    const int L  = 3;

    // ---- workspace carve-out ----
    char* ws = (char*)d_ws;
    size_t off = 0;
    auto carve = [&](size_t bytes) -> char* {
        char* p = ws + off;
        off += (bytes + 255) & ~(size_t)255;
        return p;
    };
    float* hpA  = (float*)carve((size_t)NP * 64 * 4);
    float* hpB  = (float*)carve((size_t)NP * 64 * 4);
    float* hsA  = (float*)carve((size_t)NS * 64 * 4);
    float* hsB  = (float*)carve((size_t)NS * 64 * 4);
    float* aggO = (float*)carve((size_t)NS * 64 * 4);
    float* aggH = (float*)carve((size_t)NP * 64 * 4);
    float* aggS = (float*)carve((size_t)NP * 64 * 4);
    float* invO = (float*)carve((size_t)NS * 4);
    float* invH = (float*)carve((size_t)NP * 4);
    float* invS = (float*)carve((size_t)NP * 4);
    const size_t PW32 = 1 * 4 * 32 * 16;   // halves for K<=32
    const size_t PW64 = 2 * 4 * 32 * 16;   // halves for K=64
    _Float16* pWp = (_Float16*)carve(PW32 * 2);
    _Float16* pWs = (_Float16*)carve(PW32 * 2);
    _Float16* pWl[3][3];
    _Float16* pWr0[3];
    _Float16* pWrP[3];   // Wr[l,1] + Wr[l,2] pre-summed
    for (int l = 0; l < L; ++l) {
        for (int rel = 0; rel < 3; ++rel) pWl[l][rel] = (_Float16*)carve(PW64 * 2);
        pWr0[l] = (_Float16*)carve(PW64 * 2);
        pWrP[l] = (_Float16*)carve(PW64 * 2);
    }
    (void)ws_size;

    const int B = 256;

    // ---- pack all weights into WMMA B-fragment layout ----
    pack_weights_kernel<<<cdiv(PW32, B), B, 0, stream>>>(Wp, nullptr, 32, 1, pWp);
    pack_weights_kernel<<<cdiv(PW32, B), B, 0, stream>>>(Ws, nullptr, 16, 1, pWs);
    for (int l = 0; l < L; ++l) {
        for (int rel = 0; rel < 3; ++rel)
            pack_weights_kernel<<<cdiv(PW64, B), B, 0, stream>>>(
                Wl + (size_t)(l * 3 + rel) * 64 * 64, nullptr, 64, 2, pWl[l][rel]);
        pack_weights_kernel<<<cdiv(PW64, B), B, 0, stream>>>(
            Wr + (size_t)(l * 3 + 0) * 64 * 64, nullptr, 64, 2, pWr0[l]);
        pack_weights_kernel<<<cdiv(PW64, B), B, 0, stream>>>(
            Wr + (size_t)(l * 3 + 1) * 64 * 64,
            Wr + (size_t)(l * 3 + 2) * 64 * 64, 64, 2, pWrP[l]);
    }

    // ---- degree counts (layer-invariant) -> 1/max(deg,1) ----
    zero_kernel<<<cdiv(NS, B), B, 0, stream>>>(invO, NS);
    zero_kernel<<<cdiv(NP, B), B, 0, stream>>>(invH, NP);
    zero_kernel<<<cdiv(NP, B), B, 0, stream>>>(invS, NP);
    count_kernel<<<cdiv(EO, B), B, 0, stream>>>(orb_dst,  EO, invO);
    count_kernel<<<cdiv(EH, B), B, 0, stream>>>(host_dst, EH, invH);
    count_kernel<<<cdiv(ES, B), B, 0, stream>>>(sib_dst,  ES, invS);
    invert_kernel<<<cdiv(NS, B), B, 0, stream>>>(invO, NS);
    invert_kernel<<<cdiv(NP, B), B, 0, stream>>>(invH, NP);
    invert_kernel<<<cdiv(NP, B), B, 0, stream>>>(invS, NP);

    // ---- input projections: hp = relu(x_planet@Wp + bp), hs = relu(x_star@Ws + bs)
    const int GEMM_B = 128;                       // 4 waves / block, 1 tile / wave
    const int gp = cdiv((long)NP / 16, 4);
    const int gs = cdiv((long)NS / 16, 4);
    wmma_gemm_kernel<32, 0, 0><<<gp, GEMM_B, 0, stream>>>(
        x_planet, nullptr, pWp,
        nullptr, nullptr, nullptr,
        nullptr, nullptr, nullptr,
        bp, nullptr, 1.0f, 1, hpA, NP);
    wmma_gemm_kernel<16, 0, 0><<<gs, GEMM_B, 0, stream>>>(
        x_star, nullptr, pWs,
        nullptr, nullptr, nullptr,
        nullptr, nullptr, nullptr,
        bs, nullptr, 1.0f, 1, hsA, NS);

    float* hp = hpA; float* hp_n = hpB;
    float* hs = hsA; float* hs_n = hsB;

    for (int l = 0; l < L; ++l) {
        // zero aggregation buffers
        zero_kernel<<<cdiv((long)NS * 64, B), B, 0, stream>>>(aggO, NS * 64);
        zero_kernel<<<cdiv((long)NP * 64, B), B, 0, stream>>>(aggH, NP * 64);
        zero_kernel<<<cdiv((long)NP * 64, B), B, 0, stream>>>(aggS, NP * 64);

        // gather + scatter-add (sum; mean folded into GEMM via inv-degree row scale)
        scatter_add_kernel<<<cdiv((long)EO * 16, B), B, 0, stream>>>(hp, orb_src,  orb_dst,  EO, aggO);
        scatter_add_kernel<<<cdiv((long)EH * 16, B), B, 0, stream>>>(hs, host_src, host_dst, EH, aggH);
        scatter_add_kernel<<<cdiv((long)ES * 16, B), B, 0, stream>>>(hp, sib_src,  sib_dst,  ES, aggS);

        const float* bl0 = bl + (size_t)(l * 3 + 0) * 64;
        const float* bl1 = bl + (size_t)(l * 3 + 1) * 64;
        const float* bl2 = bl + (size_t)(l * 3 + 2) * 64;

        // stars: hs_n = relu(mean(aggO)@Wl0 + bl0 + hs@Wr0)
        wmma_gemm_kernel<64, 64, 0><<<gs, GEMM_B, 0, stream>>>(
            aggO, invO, pWl[l][0],
            hs,   nullptr, pWr0[l],
            nullptr, nullptr, nullptr,
            bl0, nullptr, 1.0f, 1, hs_n, NS);

        // planets: hp_n = relu(0.5*(mean(aggH)@Wl1 + mean(aggS)@Wl2 + hp@(Wr1+Wr2) + bl1 + bl2))
        wmma_gemm_kernel<64, 64, 64><<<gp, GEMM_B, 0, stream>>>(
            aggH, invH, pWl[l][1],
            aggS, invS, pWl[l][2],
            hp,   nullptr, pWrP[l],
            bl1, bl2, 0.5f, 1, hp_n, NP);

        { float* t = hp; hp = hp_n; hp_n = t; }
        { float* t = hs; hs = hs_n; hs_n = t; }
    }

    readout_kernel<<<cdiv(NP, B), B, 0, stream>>>(hp, W1, b1, W2, b2, (float*)d_out, NP);
}
